// MultiHeadAttention_11708080849567
// MI455X (gfx1250) — compile-verified, matrix operand order checked by reading, not run on hardware
//
#include <hip/hip_runtime.h>
#include <hip/hip_bf16.h>

typedef __attribute__((ext_vector_type(16))) _Float16 v16h;
typedef __attribute__((ext_vector_type(8)))  _Float16 v8h;
typedef __attribute__((ext_vector_type(4)))  _Float16 v4h;
typedef __attribute__((ext_vector_type(2)))  _Float16 v2h;
typedef __attribute__((ext_vector_type(2)))  __fp16   v2fp;
typedef __attribute__((ext_vector_type(8)))  float    v8f;
typedef __attribute__((ext_vector_type(4)))  unsigned int v4u;
typedef __attribute__((ext_vector_type(8)))  int      v8i;
typedef __attribute__((ext_vector_type(4)))  int      v4i;

#define B_  4
#define S_  1024
#define E_  1024
#define H_  16
#define D_  64

#if __has_builtin(__builtin_amdgcn_tensor_load_to_lds) && \
    __has_builtin(__builtin_amdgcn_s_wait_tensorcnt)
#define HAS_TDM 1
#endif

// ---------------------------------------------------------------------------
__device__ __forceinline__ v8f wmma_f16(v16h a, v16h b, v8f c) {
  return __builtin_amdgcn_wmma_f32_16x16x32_f16(
      false, a, false, b, (short)0, c, false, false);
}

// One 16x32 f16 fragment row for this lane; `rowp` = start of this lane's row.
// ISA layout: lanes 0-15 hold K=0..7 & 16..23, lanes 16-31 hold K=8..15 & 24..31.
__device__ __forceinline__ v16h frag_row(const _Float16* rowp) {
  const int lane = threadIdx.x & 31;
  const int k0 = (lane >> 4) * 8;
  v8h lo = *(const v8h*)(rowp + k0);
  v8h hi = *(const v8h*)(rowp + k0 + 16);
  v16h r;
#pragma unroll
  for (int i = 0; i < 8; ++i) { r[i] = lo[i]; r[i + 8] = hi[i]; }
  return r;
}

// packed f32x4 -> f16x4 conversion (v_cvt_pk_f16_f32 when available)
__device__ __forceinline__ v4h cvt4(float4 t) {
#if __has_builtin(__builtin_amdgcn_cvt_pkrtz)
  v2fp lo = __builtin_amdgcn_cvt_pkrtz(t.x, t.y);
  v2fp hi = __builtin_amdgcn_cvt_pkrtz(t.z, t.w);
  v2h lo2 = __builtin_bit_cast(v2h, lo);
  v2h hi2 = __builtin_bit_cast(v2h, hi);
  v4h r; r[0] = lo2[0]; r[1] = lo2[1]; r[2] = hi2[0]; r[3] = hi2[1];
#else
  v4h r; r[0] = (_Float16)t.x; r[1] = (_Float16)t.y;
  r[2] = (_Float16)t.z; r[3] = (_Float16)t.w;
#endif
  return r;
}

#ifdef HAS_TDM
// TDM 2-D tile load: global f16 (row stride stride_x elems) -> LDS.
// D# packing per ISA 8.3/8.4: g0 = count|lds_addr|global_addr|type=2,
// g1 = data_size(2B)|tensor_dim0/1|tile_dim0/1|dim0_stride.
__device__ __forceinline__ void tdm_load_2d(unsigned lds_addr, const void* gptr,
                                            unsigned tile_x, unsigned tile_y,
                                            unsigned tensor_x, unsigned tensor_y,
                                            unsigned stride_x) {
  unsigned long long ga = (unsigned long long)(size_t)gptr;
  v4u g0;
  g0[0] = 1u;
  g0[1] = lds_addr;
  g0[2] = (unsigned)ga;
  g0[3] = (unsigned)((ga >> 32) & 0x01FFFFFFu) | (2u << 30);
  v8i g1;
  g1[0] = 0x00010000;                                    // data_size = 2 bytes
  g1[1] = (int)((tensor_x & 0xFFFFu) << 16);
  g1[2] = (int)((tensor_x >> 16) | ((tensor_y & 0xFFFFu) << 16));
  g1[3] = (int)((tensor_y >> 16) | (tile_x << 16));
  g1[4] = (int)tile_y;
  g1[5] = (int)stride_x;
  g1[6] = 0;
  g1[7] = 0;
  v4i z4 = {0, 0, 0, 0};
#if defined(__clang_major__) && __clang_major__ >= 23
  v8i z8 = {0, 0, 0, 0, 0, 0, 0, 0};
  __builtin_amdgcn_tensor_load_to_lds(g0, g1, z4, z4, z8, 0);
#else
  __builtin_amdgcn_tensor_load_to_lds(g0, g1, z4, z4, 0);
#endif
}
#endif

// ---------------------------------------------------------------------------
// GEMM: C[M,N] = A[M,K] * W[N,K]^T + bias[N]
//   A_IS_F16: A is f16 (ctx, TDM-loaded) vs f32 (raw inputs, convert+stage)
//   OMODE 0: f32 out [M,N]   1: f16 split-head [B,H,S,D]   2: f16 [B,H,D,S]
// Block 128 (4 waves). 64x128 C tile, wave = 32x64, K-tile 32.
// ---------------------------------------------------------------------------
template <bool A_IS_F16, int OMODE>
__global__ __launch_bounds__(128) void gemm_wmma(
    const void* __restrict__ Ap, const float* __restrict__ W,
    const float* __restrict__ bias, void* __restrict__ Cp,
    int M, int N, int K) {
  __shared__ __align__(16) _Float16 As[64 * 32];
  __shared__ __align__(16) _Float16 Bs[128 * 32];

  const int tid  = threadIdx.x;
  const int lane = tid & 31;
  const int w    = tid >> 5;
  const int m0   = blockIdx.y * 64;
  const int n0   = blockIdx.x * 128;
  const int wm   = (w & 1) * 32;
  const int wn   = (w >> 1) * 64;
  const int col  = lane & 15;
  const int hi   = lane >> 4;

  v8f acc[2][4] = {};

  const int lrow = tid >> 3;        // 0..15
  const int lcol = (tid & 7) * 4;   // 0,4,...,28

#ifdef HAS_TDM
  const unsigned ldsA = (unsigned)(size_t)&As[0];
#endif

  for (int k0 = 0; k0 < K; k0 += 32) {
    bool tdmA = false;
    if constexpr (A_IS_F16) {
#ifdef HAS_TDM
      if (w == 0) {
        tdm_load_2d(ldsA, &((const _Float16*)Ap)[(size_t)m0 * K + k0],
                    32, 64, (unsigned)K, (unsigned)M, (unsigned)K);
      }
      tdmA = true;
#else
      const _Float16* A = (const _Float16*)Ap;
#pragma unroll
      for (int rr = 0; rr < 4; ++rr) {
        const int r = lrow + rr * 16;
        *(v4h*)&As[r * 32 + lcol] =
            *(const v4h*)&A[(size_t)(m0 + r) * K + k0 + lcol];
      }
#endif
    } else {
      const float* A = (const float*)Ap;
#pragma unroll
      for (int rr = 0; rr < 4; ++rr) {
        const int r = lrow + rr * 16;
        float4 t = *(const float4*)&A[(size_t)(m0 + r) * K + k0 + lcol];
        *(v4h*)&As[r * 32 + lcol] = cvt4(t);
      }
    }
#pragma unroll
    for (int rr = 0; rr < 8; ++rr) {       // stage W: 128x32
      const int r = lrow + rr * 16;
      float4 t = *(const float4*)&W[(size_t)(n0 + r) * K + k0 + lcol];
      *(v4h*)&Bs[r * 32 + lcol] = cvt4(t);
    }
#ifdef HAS_TDM
    if (tdmA && w == 0) __builtin_amdgcn_s_wait_tensorcnt(0);
#endif
    (void)tdmA;
    __syncthreads();

    v16h a[2], b[4];
#pragma unroll
    for (int i = 0; i < 2; ++i) a[i] = frag_row(&As[(wm + i * 16 + col) * 32]);
#pragma unroll
    for (int j = 0; j < 4; ++j) b[j] = frag_row(&Bs[(wn + j * 16 + col) * 32]);
#pragma unroll
    for (int i = 0; i < 2; ++i)
#pragma unroll
      for (int j = 0; j < 4; ++j) acc[i][j] = wmma_f16(a[i], b[j], acc[i][j]);
    __syncthreads();
  }

#pragma unroll
  for (int i = 0; i < 2; ++i) {
#pragma unroll
    for (int j = 0; j < 4; ++j) {
      const int gn = n0 + wn + j * 16 + col;
      const float bv = bias[gn];
#pragma unroll
      for (int r = 0; r < 8; ++r) {
        const int gm = m0 + wm + i * 16 + r + hi * 8;
        const float val = acc[i][j][r] + bv;
        if constexpr (OMODE == 0) {
          ((float*)Cp)[(size_t)gm * N + gn] = val;
        } else if constexpr (OMODE == 1) {
          const int b_ = gm >> 10, s = gm & (S_ - 1);
          const int h = gn >> 6, d = gn & (D_ - 1);
          ((_Float16*)Cp)[((((size_t)b_ * H_ + h) * S_ + s) << 6) + d] =
              (_Float16)val;
        } else {  // transposed split-head [B,H,D,S] (for V)
          const int b_ = gm >> 10, s = gm & (S_ - 1);
          const int h = gn >> 6, d = gn & (D_ - 1);
          ((_Float16*)Cp)[(((size_t)b_ * H_ + h) * D_ + d) * S_ + s] =
              (_Float16)val;
        }
      }
    }
  }
}

// ---------------------------------------------------------------------------
// Flash-attention: grid = B*H*(S/64), block 128 (4 waves x 16 q-rows).
// Double-buffered TDM streams K [32 keys][64 d] and V^T [64 d][32 keys] tiles;
// shared-max online softmax with deferred row-sum reduction.
// ---------------------------------------------------------------------------
__global__ __launch_bounds__(128) void attn_wmma(
    const _Float16* __restrict__ Qh, const _Float16* __restrict__ Kh,
    const _Float16* __restrict__ VTh, const int* __restrict__ mask,
    _Float16* __restrict__ Ctx) {
#ifdef HAS_TDM
  __shared__ __align__(16) _Float16 Ks[2][32 * 64];   // [key][d]
  __shared__ __align__(16) _Float16 Vts[2][64 * 32];  // [d][key]
#endif
  __shared__ __align__(16) _Float16 P[4][16 * 32];

  const int tid  = threadIdx.x;
  const int lane = tid & 31;
  const int w    = tid >> 5;
  const int qb   = blockIdx.x & 15;
  const int bh   = blockIdx.x >> 4;
  const int b    = bh >> 4;
  const int col  = lane & 15;
  const int hi   = lane >> 4;

  const _Float16* Qp  = Qh  + (size_t)bh * S_ * D_;
  const _Float16* Kp  = Kh  + (size_t)bh * S_ * D_;
  const _Float16* VTp = VTh + (size_t)bh * D_ * S_;
  const int* mp = mask + b * S_;

  const int q0 = qb * 64 + w * 16;

  const _Float16* qrow = Qp + (size_t)(q0 + col) * D_;
  const v16h aq0 = frag_row(qrow);
  const v16h aq1 = frag_row(qrow + 32);

  float mrun = -1e30f;   // shared reference max for this lane-group's 8 rows
  float lrun[8];
#pragma unroll
  for (int r = 0; r < 8; ++r) lrun[r] = 0.0f;
  v8f o[4] = {};

  const int NT = S_ / 32;

#ifdef HAS_TDM
  if (w == 0) {          // prologue: tile 0 -> buffer 0
    tdm_load_2d((unsigned)(size_t)&Ks[0][0], Kp, 64, 32, 64, S_, 64);
    tdm_load_2d((unsigned)(size_t)&Vts[0][0], VTp, 32, 64, S_, D_, S_);
  }
#endif

  for (int t = 0; t < NT; ++t) {
    const int key0 = t * 32;
#ifdef HAS_TDM
    __syncthreads();     // everyone done reading buffer (t+1)&1 from tile t-1
    if (w == 0) {
      if (t + 1 < NT) {
        const int nb = (t + 1) & 1;
        tdm_load_2d((unsigned)(size_t)&Ks[nb][0],
                    Kp + (size_t)(key0 + 32) * D_, 64, 32, 64, S_, 64);
        tdm_load_2d((unsigned)(size_t)&Vts[nb][0],
                    VTp + key0 + 32, 32, 64, S_, D_, S_);
        __builtin_amdgcn_s_wait_tensorcnt(2);   // tile t complete
      } else {
        __builtin_amdgcn_s_wait_tensorcnt(0);
      }
    }
    __syncthreads();     // tile t visible to all waves
    const _Float16* Kbuf = &Ks[t & 1][0];
    const _Float16* Vbuf = &Vts[t & 1][0];
#endif

    // --- scores for both 16-key groups (4 WMMAs) ---
    v8f s0 = {}, s1 = {};
    {
#ifdef HAS_TDM
      const _Float16* kr0 = Kbuf + (size_t)col * 64;
      const _Float16* kr1 = Kbuf + (size_t)(16 + col) * 64;
#else
      const _Float16* kr0 = Kp + (size_t)(key0 + col) * D_;
      const _Float16* kr1 = Kp + (size_t)(key0 + 16 + col) * D_;
#endif
      s0 = wmma_f16(aq0, frag_row(kr0), s0);
      s0 = wmma_f16(aq1, frag_row(kr0 + 32), s0);
      s1 = wmma_f16(aq0, frag_row(kr1), s1);
      s1 = wmma_f16(aq1, frag_row(kr1 + 32), s1);
    }

    const bool dead0 = (mp[key0 + col] == 0);
    const bool dead1 = (mp[key0 + 16 + col] == 0);

    // --- masked, scaled scores; single shared max per lane-group ---
    float a8[8], c8[8];
#pragma unroll
    for (int r = 0; r < 8; ++r) {
      a8[r] = dead0 ? -3e30f : s0[r] * 0.125f;   // 1/sqrt(64)
      c8[r] = dead1 ? -3e30f : s1[r] * 0.125f;
    }
    float tmax = fmaxf(a8[0], c8[0]);
#pragma unroll
    for (int r = 1; r < 8; ++r) tmax = fmaxf(tmax, fmaxf(a8[r], c8[r]));
    tmax = fmaxf(tmax, __shfl_xor(tmax, 1, 32));
    tmax = fmaxf(tmax, __shfl_xor(tmax, 2, 32));
    tmax = fmaxf(tmax, __shfl_xor(tmax, 4, 32));
    tmax = fmaxf(tmax, __shfl_xor(tmax, 8, 32));

    const float mnew = fmaxf(mrun, tmax);
    const float corr = __expf(mrun - mnew);      // uniform across lane-group
    mrun = mnew;

#pragma unroll
    for (int r = 0; r < 8; ++r) {
      const float p0 = __expf(a8[r] - mnew);     // masked -> underflows to 0
      const float p1 = __expf(c8[r] - mnew);
      lrun[r] = lrun[r] * corr + (p0 + p1);      // per-lane partial row sum
#pragma unroll
      for (int dt = 0; dt < 4; ++dt) o[dt][r] *= corr;
      P[w][(r + hi * 8) * 32 + col] = (_Float16)p0;   // C-layout -> A-layout
      P[w][(r + hi * 8) * 32 + 16 + col] = (_Float16)p1;
    }

    // wave-private LDS RAW (same-wave DS in-order; split-counter wait)
    asm volatile("s_wait_dscnt 0" ::: "memory");

    v16h ap = frag_row(&P[w][col * 32]);
#pragma unroll
    for (int dt = 0; dt < 4; ++dt) {
#ifdef HAS_TDM
      v16h bv = frag_row(Vbuf + (size_t)(dt * 16 + col) * 32);
#else
      v16h bv = frag_row(VTp + (size_t)(dt * 16 + col) * S_ + key0);
#endif
      o[dt] = wmma_f16(ap, bv, o[dt]);
    }
  }

  // deferred row-sum reduction (once), normalize, write ctx f16 [B,S,E]
  const int h = bh & 15;
#pragma unroll
  for (int r = 0; r < 8; ++r) {
    float psum = lrun[r];
    psum += __shfl_xor(psum, 1, 32);
    psum += __shfl_xor(psum, 2, 32);
    psum += __shfl_xor(psum, 4, 32);
    psum += __shfl_xor(psum, 8, 32);
    const float inv = psum > 0.0f ? 1.0f / psum : 0.0f;
    const int q = q0 + r + hi * 8;
    const size_t rowoff = (size_t)(b * S_ + q) * E_ + h * D_;
#pragma unroll
    for (int dt = 0; dt < 4; ++dt) {
      Ctx[rowoff + dt * 16 + col] = (_Float16)(o[dt][r] * inv);
    }
  }
}

// ---------------------------------------------------------------------------
extern "C" void kernel_launch(void* const* d_in, const int* in_sizes, int n_in,
                              void* d_out, int out_size, void* d_ws,
                              size_t ws_size, hipStream_t stream) {
  const float* q    = (const float*)d_in[0];
  const float* k    = (const float*)d_in[1];
  const float* v    = (const float*)d_in[2];
  const int*   mask = (const int*)d_in[3];
  const float* Wq   = (const float*)d_in[4];
  const float* bq   = (const float*)d_in[5];
  const float* Wk   = (const float*)d_in[6];
  const float* bk   = (const float*)d_in[7];
  const float* Wo   = (const float*)d_in[8];
  const float* bo   = (const float*)d_in[9];

  const size_t headElems = (size_t)B_ * H_ * S_ * D_;  // 8 MB f16 each
  _Float16* wsq = (_Float16*)d_ws;                     // [B,H,S,D]
  _Float16* wsk = wsq + headElems;                     // [B,H,S,D]
  _Float16* wsv = wsk + headElems;                     // [B,H,D,S] (transposed)
  _Float16* wsc = wsv + headElems;                     // ctx [B,S,E]

  const int M = B_ * S_, N = E_, K = E_;
  dim3 grid(N / 128, M / 64), blk(128);

  gemm_wmma<false, 1><<<grid, blk, 0, stream>>>(q, Wq, bq, wsq, M, N, K);
  gemm_wmma<false, 1><<<grid, blk, 0, stream>>>(k, Wk, bk, wsk, M, N, K);
  gemm_wmma<false, 2><<<grid, blk, 0, stream>>>(v, Wk, bk, wsv, M, N, K);

  attn_wmma<<<B_ * H_ * (S_ / 64), blk, 0, stream>>>(wsq, wsk, wsv, mask, wsc);

  gemm_wmma<true, 0><<<grid, blk, 0, stream>>>(wsc, Wo, bo, (float*)d_out,
                                               M, N, K);
}
